// _LATTICE_51668456571002
// MI455X (gfx1250) — matrix operand order repaired and checked by reading, not compiled, not run on hardware
//
#include <hip/hip_runtime.h>
#include <math.h>

#define N_USERS 19968
#define N_ITEMS 6144
#define EMBED   64
#define V_DIM   4096
#define T_DIM   384
#define KNN     10
#define NNZE    600000
#define N_NODES (N_USERS + N_ITEMS)
#define LAMBDA  0.9f
#define SIM_WAVES 4
#define GW 4   // waves per block in the streaming GEMMs

typedef __attribute__((ext_vector_type(16))) __bf16   v16bf;
typedef __attribute__((ext_vector_type(8)))  float    v8f;
typedef __attribute__((ext_vector_type(8)))  unsigned v8u;

// ---------------- WMMA helpers (CDNA5 wave32 layouts, ISA 7.12.2) ----------

__device__ __forceinline__ unsigned pack2(float lo, float hi) {
  __bf16 a = (__bf16)lo, b = (__bf16)hi;
  unsigned short ua = __builtin_bit_cast(unsigned short, a);
  unsigned short ub = __builtin_bit_cast(unsigned short, b);
  return (unsigned)ua | ((unsigned)ub << 16);
}

__device__ __forceinline__ v8f wmma_bf16(v16bf a, v16bf b, v8f c) {
  return __builtin_amdgcn_wmma_f32_16x16x32_bf16(
      /*neg_a=*/false, a, /*neg_b=*/false, b,
      /*c_mod=*/(short)0, c, /*reuse_a=*/false, /*reuse_b=*/false);
}

// A fragment (16x32, M x K) from fp32 LDS tile a[16][ld], scaled+converted to bf16.
__device__ __forceinline__ v16bf a_frag_f32(const float* a, int ld, float scale) {
  int l = threadIdx.x & 31;
  int m = l & 15;
  int kb = (l & 16) ? 8 : 0;
  v8u u;
#pragma unroll
  for (int r = 0; r < 8; ++r) {
    int k = (r < 4) ? (kb + 2 * r) : (16 + kb + 2 * (r - 4));
    u[r] = pack2(a[m * ld + k] * scale, a[m * ld + k + 1] * scale);
  }
  return __builtin_bit_cast(v16bf, u);
}

// B fragment (32x16, K x N) from fp32 LDS tile b[32][ld].
__device__ __forceinline__ v16bf b_frag_f32(const float* b, int ld) {
  int l = threadIdx.x & 31;
  int n = l & 15;
  int kb = (l & 16) ? 8 : 0;
  v8u u;
#pragma unroll
  for (int r = 0; r < 8; ++r) {
    int k = (r < 4) ? (kb + 2 * r) : (16 + kb + 2 * (r - 4));
    u[r] = pack2(b[k * ld + n], b[(k + 1) * ld + n]);
  }
  return __builtin_bit_cast(v16bf, u);
}

// A fragment from bf16 LDS tile x[16][64] at K-offset ko (b32 LDS reads of bf16 pairs)
__device__ __forceinline__ v16bf a_frag_bf(const __bf16* x, int ko) {
  int l = threadIdx.x & 31;
  int m = l & 15;
  int kb = (l & 16) ? 8 : 0;
  v8u u;
#pragma unroll
  for (int r = 0; r < 8; ++r) {
    int k = ko + ((r < 4) ? (kb + 2 * r) : (16 + kb + 2 * (r - 4)));
    u[r] = *(const unsigned*)(x + m * 64 + k);
  }
  return __builtin_bit_cast(v16bf, u);
}

// B fragment where B[k][n] = x[n][k] (x = 16 rows of Xn, row-major [16][64] bf16)
__device__ __forceinline__ v16bf b_frag_bf_T(const __bf16* x, int ko) {
  int l = threadIdx.x & 31;
  int n = l & 15;
  int kb = (l & 16) ? 8 : 0;
  v8u u;
#pragma unroll
  for (int r = 0; r < 8; ++r) {
    int k = ko + ((r < 4) ? (kb + 2 * r) : (16 + kb + 2 * (r - 4)));
    u[r] = *(const unsigned*)(x + n * 64 + k);
  }
  return __builtin_bit_cast(v16bf, u);
}

__device__ __forceinline__ void modal_w(const float* mw, float& w0, float& w1) {
  float m  = fmaxf(mw[0], mw[1]);
  float e0 = __expf(mw[0] - m), e1 = __expf(mw[1] - m);
  float inv = 1.0f / (e0 + e1);
  w0 = e0 * inv; w1 = e1 * inv;
}

// ---------------- K1: feats = F(MxK) @ W(Kx64) + bias -----------------------
// grid M/64, block 128 (4 waves). Each wave owns one 16-row stripe; the W tile
// is staged once per block. Each F byte is read exactly once.
__global__ void gemm_feats_kernel(const float* __restrict__ F, const float* __restrict__ W,
                                  const float* __restrict__ bias, float* __restrict__ out,
                                  int K) {
  __shared__ __align__(16) float At[GW][16 * 32];
  __shared__ __align__(16) float Bt[32 * 64];
  int lane = threadIdx.x & 31;
  int wv   = threadIdx.x >> 5;
  int m0 = (blockIdx.x * GW + wv) * 16;
  v8f c[4] = {};
  for (int k0 = 0; k0 < K; k0 += 32) {
    __syncthreads();
#pragma unroll
    for (int t = lane; t < 128; t += 32) {  // per-wave A: 16x32 via b128
      int r = t >> 3, c4 = (t & 7) * 4;
      *(float4*)&At[wv][r * 32 + c4] = *(const float4*)&F[(size_t)(m0 + r) * K + k0 + c4];
    }
    {  // shared B: 32x64 via b128, all 128 threads
      int t = threadIdx.x;
#pragma unroll
      for (int i = 0; i < 4; ++i, t += 128) {
        int r = t >> 4, c4 = (t & 15) * 4;
        *(float4*)&Bt[r * 64 + c4] = *(const float4*)&W[(size_t)(k0 + r) * EMBED + c4];
      }
    }
    __syncthreads();
    v16bf a = a_frag_f32(At[wv], 32, 1.0f);
#pragma unroll
    for (int j = 0; j < 4; ++j) {
      v16bf b = b_frag_f32(Bt + j * 16, 64);
      c[j] = wmma_bf16(a, b, c[j]);
    }
  }
  int n = lane & 15, mb = (lane & 16) ? 8 : 0;
#pragma unroll
  for (int j = 0; j < 4; ++j) {
    float bs = bias[j * 16 + n];
#pragma unroll
    for (int r = 0; r < 8; ++r)
      out[(size_t)(m0 + mb + r) * EMBED + j * 16 + n] = c[j][r] + bs;
  }
}

// ---------------- K2: row-normalize feats -> bf16 ---------------------------
__global__ void rownorm_bf16_kernel(const float* __restrict__ feats, __bf16* __restrict__ xn) {
  int gid = blockIdx.x * blockDim.x + threadIdx.x;
  int row = gid >> 5, lane = gid & 31;
  if (row >= N_ITEMS) return;
  float x0 = feats[(size_t)row * 64 + lane];
  float x1 = feats[(size_t)row * 64 + 32 + lane];
  float ss = x0 * x0 + x1 * x1;
#pragma unroll
  for (int off = 16; off; off >>= 1) ss += __shfl_xor(ss, off, 32);
  float inv = rsqrtf(ss);
  xn[(size_t)row * 64 + lane]      = (__bf16)(x0 * inv);
  xn[(size_t)row * 64 + 32 + lane] = (__bf16)(x1 * inv);
}

// ---------------- K3: fused  S = Xn @ Xn^T  +  streaming top-10 -------------
// grid 96, block 128 (4 waves); each wave owns one 16-row stripe, the block
// shares the streamed 16-column tile in LDS.
__global__ void sim_topk_kernel(const __bf16* __restrict__ xn,
                                float* __restrict__ topv, int* __restrict__ topi) {
  __shared__ __align__(16) __bf16 xi[SIM_WAVES][16 * 64];
  __shared__ __align__(16) __bf16 xj[16 * 64];
  __shared__ float scores[SIM_WAVES][16 * 16];
  __shared__ float tv[SIM_WAVES][16 * KNN];
  __shared__ int   ti[SIM_WAVES][16 * KNN];
  int lane = threadIdx.x & 31;
  int wv   = threadIdx.x >> 5;
  int m0 = (blockIdx.x * SIM_WAVES + wv) * 16;

  {  // own stripe -> LDS (b128)
    const uint4* src = (const uint4*)(xn + (size_t)m0 * 64);
    uint4* dst = (uint4*)xi[wv];
#pragma unroll
    for (int t = lane; t < 128; t += 32) dst[t] = src[t];
  }
  if (lane < 16) {
#pragma unroll
    for (int q = 0; q < KNN; ++q) { tv[wv][lane * KNN + q] = -3.0e38f; ti[wv][lane * KNN + q] = 0; }
  }
  __syncthreads();
  v16bf a0 = a_frag_bf(xi[wv], 0);
  v16bf a1 = a_frag_bf(xi[wv], 32);

  for (int j0 = 0; j0 < N_ITEMS; j0 += 16) {
    __syncthreads();  // protect xj reuse across all 4 waves
    {
      const uint4* src = (const uint4*)(xn + (size_t)j0 * 64);
      uint4* dst = (uint4*)xj;
      int t = threadIdx.x;
      if (t < 128) dst[t] = src[t];
    }
    __syncthreads();
    v16bf b0 = b_frag_bf_T(xj, 0);
    v16bf b1 = b_frag_bf_T(xj, 32);
    v8f c = {};
    c = wmma_bf16(a0, b0, c);
    c = wmma_bf16(a1, b1, c);
    int n = lane & 15, mb = (lane & 16) ? 8 : 0;
#pragma unroll
    for (int r = 0; r < 8; ++r) scores[wv][(mb + r) * 16 + n] = c[r];
    // wave-local RAW on scores -> compiler inserts s_wait_dscnt; no barrier needed
    if (lane < 16) {  // each lane owns one row's sorted top-10
      float* v  = &tv[wv][lane * KNN];
      int*   ix = &ti[wv][lane * KNN];
#pragma unroll
      for (int q = 0; q < 16; ++q) {
        float s = scores[wv][lane * 16 + q];
        if (s > v[KNN - 1]) {
          int p = KNN - 1;
          while (p > 0 && v[p - 1] < s) { v[p] = v[p - 1]; ix[p] = ix[p - 1]; --p; }
          v[p] = s; ix[p] = j0 + q;
        }
      }
    }
  }
  __syncthreads();
  if (lane < 16) {
#pragma unroll
    for (int q = 0; q < KNN; ++q) {
      topv[(size_t)(m0 + lane) * KNN + q] = tv[wv][lane * KNN + q];
      topi[(size_t)(m0 + lane) * KNN + q] = ti[wv][lane * KNN + q];
    }
  }
}

// ---------------- K4: combine modal top-k rows + Laplacian degree ----------
__global__ void combine_topk_kernel(const float* __restrict__ iv, const int* __restrict__ ii,
                                    const float* __restrict__ tvv, const int* __restrict__ tii,
                                    const float* __restrict__ mw,
                                    float* __restrict__ cval, int* __restrict__ ccol,
                                    float* __restrict__ dinv) {
  int i = blockIdx.x * blockDim.x + threadIdx.x;
  if (i >= N_ITEMS) return;
  float w0, w1; modal_w(mw, w0, w1);
  float s = 0.f;
#pragma unroll
  for (int q = 0; q < KNN; ++q) {
    float v = w0 * iv[(size_t)i * KNN + q];
    cval[(size_t)i * 20 + q] = v; ccol[(size_t)i * 20 + q] = ii[(size_t)i * KNN + q];
    s += v;
  }
#pragma unroll
  for (int q = 0; q < KNN; ++q) {
    float v = w1 * tvv[(size_t)i * KNN + q];
    cval[(size_t)i * 20 + KNN + q] = v; ccol[(size_t)i * 20 + KNN + q] = tii[(size_t)i * KNN + q];
    s += v;
  }
  dinv[i] = (s > 0.f) ? rsqrtf(s) : 0.f;  // d^-0.5, inf -> 0
}

// ---------------- K5: dense  h = lambda*(w0*Ai + w1*At) @ emb  (WMMA) -------
// grid 96, block 128 (4 waves). Each wave owns one 16-row stripe of both
// adjacencies (each byte read once); the emb tile is staged once per block.
__global__ void dense_adj_gemm_kernel(const float* __restrict__ Ai, const float* __restrict__ At,
                                      const float* __restrict__ emb, const float* __restrict__ mw,
                                      float* __restrict__ h) {
  __shared__ __align__(16) float Ta[GW][16 * 32];
  __shared__ __align__(16) float Tb[GW][16 * 32];
  __shared__ __align__(16) float Te[32 * 64];
  int lane = threadIdx.x & 31;
  int wv   = threadIdx.x >> 5;
  int m0 = (blockIdx.x * GW + wv) * 16;
  float w0, w1; modal_w(mw, w0, w1);
  float s0 = LAMBDA * w0, s1 = LAMBDA * w1;
  v8f c[4] = {};
  for (int k0 = 0; k0 < N_ITEMS; k0 += 32) {
    __syncthreads();
#pragma unroll
    for (int t = lane; t < 128; t += 32) {  // per-wave A tiles, b128
      int r = t >> 3, c4 = (t & 7) * 4;
      size_t off = (size_t)(m0 + r) * N_ITEMS + k0 + c4;
      *(float4*)&Ta[wv][r * 32 + c4] = *(const float4*)&Ai[off];
      *(float4*)&Tb[wv][r * 32 + c4] = *(const float4*)&At[off];
    }
    {  // shared emb tile: 32x64, b128 (L2-resident), all 128 threads
      int t = threadIdx.x;
#pragma unroll
      for (int i = 0; i < 4; ++i, t += 128) {
        int r = t >> 4, c4 = (t & 15) * 4;
        *(float4*)&Te[r * 64 + c4] = *(const float4*)&emb[(size_t)(k0 + r) * EMBED + c4];
      }
    }
    if (k0 + 32 < N_ITEMS) {  // stream hint for next A tiles -> global_prefetch_b8
      size_t pf = (size_t)(m0 + (lane >> 1)) * N_ITEMS + k0 + 32 + (lane & 1) * 16;
      __builtin_prefetch(&Ai[pf], 0, 0);
      __builtin_prefetch(&At[pf], 0, 0);
    }
    __syncthreads();
    v16bf ai = a_frag_f32(Ta[wv], 32, s0);
    v16bf at = a_frag_f32(Tb[wv], 32, s1);
#pragma unroll
    for (int j = 0; j < 4; ++j) {
      v16bf b = b_frag_f32(Te + j * 16, 64);
      c[j] = wmma_bf16(ai, b, c[j]);
      c[j] = wmma_bf16(at, b, c[j]);
    }
  }
  int n = lane & 15, mb = (lane & 16) ? 8 : 0;
#pragma unroll
  for (int j = 0; j < 4; ++j)
#pragma unroll
    for (int r = 0; r < 8; ++r)
      h[(size_t)(m0 + mb + r) * EMBED + j * 16 + n] = c[j][r];
}

// ---------------- K6: sparse  h += (1-lambda) * learned_adj @ emb -----------
__global__ void spmm_learned_kernel(const float* __restrict__ cval, const int* __restrict__ ccol,
                                    const float* __restrict__ dinv, const float* __restrict__ emb,
                                    float* __restrict__ h) {
  int gid = blockIdx.x * blockDim.x + threadIdx.x;
  int row = gid >> 5, lane = gid & 31;
  if (row >= N_ITEMS) return;
  float di = dinv[row];
  float acc0 = 0.f, acc1 = 0.f;
#pragma unroll
  for (int q = 0; q < 20; ++q) {
    float v = cval[(size_t)row * 20 + q];
    int   c = ccol[(size_t)row * 20 + q];
    float w = (1.0f - LAMBDA) * v * di * dinv[c];
    acc0 += w * emb[(size_t)c * 64 + lane];
    acc1 += w * emb[(size_t)c * 64 + 32 + lane];
  }
  h[(size_t)row * 64 + lane]      += acc0;
  h[(size_t)row * 64 + 32 + lane] += acc1;
}

// ---------------- K7: UI graph propagation ---------------------------------
__global__ void concat_ego_kernel(const float* __restrict__ ue, const float* __restrict__ ie,
                                  float* __restrict__ ego, float* __restrict__ macc) {
  int idx = blockIdx.x * blockDim.x + threadIdx.x;
  if (idx >= N_NODES * EMBED) return;
  float v = (idx < N_USERS * EMBED) ? ue[idx] : ie[idx - N_USERS * EMBED];
  ego[idx]  = v;
  macc[idx] = v;
}

__global__ void fill_zero_kernel(float* __restrict__ p, int n) {
  int idx = blockIdx.x * blockDim.x + threadIdx.x;
  if (idx < n) p[idx] = 0.f;
}

// 32 lanes per edge, 2 dims per lane (all L2-resident atomics)
__global__ void scatter_edges_kernel(const int* __restrict__ ind, const float* __restrict__ vals,
                                     const float* __restrict__ ein, float* __restrict__ eout) {
  long long t = (long long)blockIdx.x * blockDim.x + threadIdx.x;
  if (t >= (long long)NNZE * 32) return;
  int e = (int)(t >> 5), d = (int)(t & 31);
  int s  = ind[e];           // adj_indices[0][e] = src
  int ds = ind[NNZE + e];    // adj_indices[1][e] = dst
  float v = vals[e];
  atomicAdd(&eout[(size_t)s * EMBED + d],      v * ein[(size_t)ds * EMBED + d]);
  atomicAdd(&eout[(size_t)s * EMBED + 32 + d], v * ein[(size_t)ds * EMBED + 32 + d]);
}

__global__ void axpy_kernel(float* __restrict__ acc, const float* __restrict__ x, int n) {
  int idx = blockIdx.x * blockDim.x + threadIdx.x;
  if (idx < n) acc[idx] += x[idx];
}

// ---------------- K8: finalize ---------------------------------------------
__global__ void finalize_users_kernel(const float* __restrict__ macc, float* __restrict__ out) {
  int idx = blockIdx.x * blockDim.x + threadIdx.x;
  if (idx < N_USERS * EMBED) out[idx] = macc[idx] * (1.0f / 3.0f);
}

__global__ void finalize_items_kernel(const float* __restrict__ macc, const float* __restrict__ h,
                                      float* __restrict__ out) {
  int gid = blockIdx.x * blockDim.x + threadIdx.x;
  int row = gid >> 5, lane = gid & 31;
  if (row >= N_ITEMS) return;
  float x0 = h[(size_t)row * 64 + lane];
  float x1 = h[(size_t)row * 64 + 32 + lane];
  float ss = x0 * x0 + x1 * x1;
#pragma unroll
  for (int off = 16; off; off >>= 1) ss += __shfl_xor(ss, off, 32);
  float inv = 1.0f / fmaxf(sqrtf(ss), 1e-12f);
  const float* mrow = macc + (size_t)(N_USERS + row) * EMBED;
  float* orow = out + (size_t)N_USERS * EMBED + (size_t)row * EMBED;
  orow[lane]      = mrow[lane]      * (1.0f / 3.0f) + x0 * inv;
  orow[32 + lane] = mrow[32 + lane] * (1.0f / 3.0f) + x1 * inv;
}

// ---------------- host ------------------------------------------------------
extern "C" void kernel_launch(void* const* d_in, const int* in_sizes, int n_in,
                              void* d_out, int out_size, void* d_ws, size_t ws_size,
                              hipStream_t stream) {
  (void)in_sizes; (void)n_in; (void)out_size; (void)ws_size;
  const float* user_emb   = (const float*)d_in[0];
  const float* item_emb   = (const float*)d_in[1];
  const float* image_feat = (const float*)d_in[2];
  const float* text_feat  = (const float*)d_in[3];
  const float* img_w      = (const float*)d_in[4];
  const float* img_b      = (const float*)d_in[5];
  const float* txt_w      = (const float*)d_in[6];
  const float* txt_b      = (const float*)d_in[7];
  const float* mw         = (const float*)d_in[8];
  const float* img_orig   = (const float*)d_in[9];
  const float* txt_orig   = (const float*)d_in[10];
  const int*   adj_ind    = (const int*)d_in[11];
  const float* adj_val    = (const float*)d_in[12];
  float* out = (float*)d_out;

  char* ws = (char*)d_ws;
  size_t o = 0;
  auto alloc = [&](size_t bytes) { size_t r = o; o += (bytes + 255) & ~(size_t)255; return r; };
  float*  feats_img = (float*)(ws + alloc((size_t)N_ITEMS * 64 * 4));
  float*  feats_txt = (float*)(ws + alloc((size_t)N_ITEMS * 64 * 4));
  __bf16* xn_img    = (__bf16*)(ws + alloc((size_t)N_ITEMS * 64 * 2));
  __bf16* xn_txt    = (__bf16*)(ws + alloc((size_t)N_ITEMS * 64 * 2));
  float*  topv_i    = (float*)(ws + alloc((size_t)N_ITEMS * KNN * 4));
  int*    topi_i    = (int*)  (ws + alloc((size_t)N_ITEMS * KNN * 4));
  float*  topv_t    = (float*)(ws + alloc((size_t)N_ITEMS * KNN * 4));
  int*    topi_t    = (int*)  (ws + alloc((size_t)N_ITEMS * KNN * 4));
  float*  cval      = (float*)(ws + alloc((size_t)N_ITEMS * 20 * 4));
  int*    ccol      = (int*)  (ws + alloc((size_t)N_ITEMS * 20 * 4));
  float*  dinv      = (float*)(ws + alloc((size_t)N_ITEMS * 4));
  float*  hbuf      = (float*)(ws + alloc((size_t)N_ITEMS * 64 * 4));
  float*  ego0      = (float*)(ws + alloc((size_t)N_NODES * 64 * 4));
  float*  egoA      = (float*)(ws + alloc((size_t)N_NODES * 64 * 4));
  float*  egoB      = (float*)(ws + alloc((size_t)N_NODES * 64 * 4));
  float*  macc      = (float*)(ws + alloc((size_t)N_NODES * 64 * 4));

  const int nvec = N_NODES * EMBED;

  // 1) modal feature transforms (WMMA bf16; 4 waves/block, F read once)
  gemm_feats_kernel<<<N_ITEMS / (16 * GW), 32 * GW, 0, stream>>>(image_feat, img_w, img_b,
                                                                 feats_img, V_DIM);
  gemm_feats_kernel<<<N_ITEMS / (16 * GW), 32 * GW, 0, stream>>>(text_feat, txt_w, txt_b,
                                                                 feats_txt, T_DIM);
  // 2) row-normalize -> bf16
  rownorm_bf16_kernel<<<(N_ITEMS * 32) / 256, 256, 0, stream>>>(feats_img, xn_img);
  rownorm_bf16_kernel<<<(N_ITEMS * 32) / 256, 256, 0, stream>>>(feats_txt, xn_txt);
  // 3) fused similarity GEMM + streaming top-10 (never materializes 151MB S)
  sim_topk_kernel<<<N_ITEMS / (16 * SIM_WAVES), 32 * SIM_WAVES, 0, stream>>>(xn_img, topv_i, topi_i);
  sim_topk_kernel<<<N_ITEMS / (16 * SIM_WAVES), 32 * SIM_WAVES, 0, stream>>>(xn_txt, topv_t, topi_t);
  // 4) combine modal graphs, Laplacian degree^-1/2
  combine_topk_kernel<<<(N_ITEMS + 255) / 256, 256, 0, stream>>>(topv_i, topi_i, topv_t, topi_t,
                                                                 mw, cval, ccol, dinv);
  // 5) h = lambda*(w0*img_orig + w1*txt_orig) @ emb  (dense; 384 waves streaming)
  dense_adj_gemm_kernel<<<N_ITEMS / (16 * GW), 32 * GW, 0, stream>>>(img_orig, txt_orig,
                                                                     item_emb, mw, hbuf);
  // 6) h += (1-lambda) * learned_adj @ emb   (sparse, <=20 nnz/row)
  spmm_learned_kernel<<<(N_ITEMS * 32) / 256, 256, 0, stream>>>(cval, ccol, dinv, item_emb, hbuf);
  // 7) UI propagation: ego^0, two scatter layers, running mean-sum
  concat_ego_kernel<<<(nvec + 255) / 256, 256, 0, stream>>>(user_emb, item_emb, ego0, macc);
  fill_zero_kernel<<<(nvec + 255) / 256, 256, 0, stream>>>(egoA, nvec);
  {
    long long tot = (long long)NNZE * 32;
    int blocks = (int)((tot + 255) / 256);
    scatter_edges_kernel<<<blocks, 256, 0, stream>>>(adj_ind, adj_val, ego0, egoA);
    axpy_kernel<<<(nvec + 255) / 256, 256, 0, stream>>>(macc, egoA, nvec);
    fill_zero_kernel<<<(nvec + 255) / 256, 256, 0, stream>>>(egoB, nvec);
    scatter_edges_kernel<<<blocks, 256, 0, stream>>>(adj_ind, adj_val, egoA, egoB);
    axpy_kernel<<<(nvec + 255) / 256, 256, 0, stream>>>(macc, egoB, nvec);
  }
  // 8) outputs: u_g then i_g (mean/3 + h/||h||)
  finalize_users_kernel<<<(N_USERS * EMBED + 255) / 256, 256, 0, stream>>>(macc, out);
  finalize_items_kernel<<<(N_ITEMS * 32) / 256, 256, 0, stream>>>(macc, hbuf, out);
}
